// BASE_MAMBA2_14018773254564
// MI455X (gfx1250) — compile-verified, matrix operand order checked by reading, not run on hardware
//
#include <hip/hip_runtime.h>
#include <hip/hip_bf16.h>
#include <math.h>

#define D_MODEL   1024
#define D_INNER   2048
#define D_STATE   64
#define HEADDIM   64
#define NHEADS    32
#define CONV_DIM  2176     // D_INNER + 2*D_STATE
#define D_IN_PROJ 4256     // 2*D_INNER + 2*D_STATE + NHEADS
#define BATCH     8
#define SEQLEN    2048
#define M_TOTAL   (BATCH * SEQLEN)   // 16384
#define QCHUNK    64
#define NCHUNK    (SEQLEN / QCHUNK)  // 32
#define EPSF      1e-5f

typedef __attribute__((ext_vector_type(16))) _Float16 v16h;
typedef __attribute__((ext_vector_type(8)))  float    v8f;

__device__ __forceinline__ float silu_f(float x) { return x / (1.f + __expf(-x)); }

// 64x64x64 GEMM helper built on v_wmma_f32_16x16x32_f16.
// aElem(r, k): A element at tile-row r (0..15), contraction index k (0..63)
// bElem(k, c): B element at contraction index k, tile-col c (0..15)
template <typename FA, typename FB>
__device__ __forceinline__ v8f mm64(FA aElem, FB bElem, v8f acc,
                                    const int lid, const int kbase) {
#pragma unroll
    for (int ks = 0; ks < 2; ++ks) {
        v16h af, bf;
#pragma unroll
        for (int j = 0; j < 16; ++j) {
            const int k = ks * 32 + kbase + ((j < 8) ? j : (8 + j));
            af[j] = (_Float16)aElem(lid, k);
            bf[j] = (_Float16)bElem(k, lid);
        }
        acc = __builtin_amdgcn_wmma_f32_16x16x32_f16(false, af, false, bf,
                                                     (short)0, acc, false, false);
    }
    return acc;
}

// ---------------------------------------------------------------------------
// 1) Wf[k][n] = (Wproj @ Win)[k][n], k padded 20->32 with zeros, stored f16.
// ---------------------------------------------------------------------------
__global__ void k_fuse_weights(const float* __restrict__ Wproj,
                               const float* __restrict__ bproj,
                               const float* __restrict__ Win,
                               _Float16* __restrict__ Wf,
                               float* __restrict__ bf) {
    const int n = blockIdx.x;
    const int t = threadIdx.x;
    float acc = 0.f;
    if (t < 20) {
        const float* wp = Wproj + t * D_MODEL;
        for (int d = 0; d < D_MODEL; ++d)
            acc += wp[d] * Win[(size_t)d * D_IN_PROJ + n];
    }
    Wf[(size_t)t * D_IN_PROJ + n] = (_Float16)acc;
    if (t == 0) {
        float ba = 0.f;
        for (int d = 0; d < D_MODEL; ++d)
            ba += bproj[d] * Win[(size_t)d * D_IN_PROJ + n];
        bf[n] = ba;
    }
}

// ---------------------------------------------------------------------------
// 2) zxbcdt = x^T @ Wf + bf  (one v_wmma per 16x16 tile, K padded 20->32)
// ---------------------------------------------------------------------------
__global__ void __launch_bounds__(256)
k_gemm_in(const float* __restrict__ x, const _Float16* __restrict__ Wf,
          const float* __restrict__ bf, float* __restrict__ zx) {
    const int lane = threadIdx.x & 31;
    const int wave = threadIdx.x >> 5;
    const int n0   = blockIdx.x * 16;
    const int m0   = (blockIdx.y * 8 + wave) * 16;
    const int half = lane >> 4;
    const int lid  = lane & 15;
    const int kbase = half * 8;

    const int mrow = m0 + lid;
    const int bidx = mrow >> 11;
    const int lpos = mrow & (SEQLEN - 1);
    const float* xb = x + (size_t)bidx * 20 * SEQLEN + lpos;
    v16h afrag, bfrag;
#pragma unroll
    for (int j = 0; j < 16; ++j) {
        const int k = kbase + ((j < 8) ? j : (8 + j));
        afrag[j] = (_Float16)((k < 20) ? xb[(size_t)k * SEQLEN] : 0.f);
    }
    const _Float16* wb = Wf + (n0 + lid);
#pragma unroll
    for (int j = 0; j < 16; ++j) {
        const int k = kbase + ((j < 8) ? j : (8 + j));
        bfrag[j] = wb[(size_t)k * D_IN_PROJ];
    }
    v8f c = {};
    c = __builtin_amdgcn_wmma_f32_16x16x32_f16(false, afrag, false, bfrag,
                                               (short)0, c, false, false);
    const float bias = bf[n0 + lid];
#pragma unroll
    for (int r = 0; r < 8; ++r) {
        const int m = m0 + r + half * 8;
        zx[(size_t)m * D_IN_PROJ + n0 + lid] = c[r] + bias;
    }
}

// ---------------------------------------------------------------------------
// 3) causal depthwise conv (4 taps) + bias + SiLU over xBC channels
// ---------------------------------------------------------------------------
__global__ void k_conv_silu(const float* __restrict__ zx,
                            const float* __restrict__ conv_w,
                            const float* __restrict__ conv_b,
                            float* __restrict__ xc) {
    const int c = blockIdx.x * 256 + threadIdx.x;
    const int m = blockIdx.y;
    if (c >= CONV_DIM) return;
    const int l = m & (SEQLEN - 1);
    float acc = conv_b[c];
#pragma unroll
    for (int k = 0; k < 4; ++k) {
        const int ll = l - 3 + k;
        if (ll >= 0)
            acc += conv_w[k * CONV_DIM + c] *
                   zx[(size_t)(m - 3 + k) * D_IN_PROJ + D_INNER + c];
    }
    xc[(size_t)m * CONV_DIM + c] = silu_f(acc);
}

// ---------------------------------------------------------------------------
// 4) dt = softplus(dt_raw + dt_bias)
// ---------------------------------------------------------------------------
__global__ void k_dt(const float* __restrict__ zx,
                     const float* __restrict__ dt_bias,
                     float* __restrict__ dtv) {
    const int i = blockIdx.x * 256 + threadIdx.x;
    if (i >= M_TOTAL * NHEADS) return;
    const int m = i >> 5;
    const int h = i & 31;
    const float raw = zx[(size_t)m * D_IN_PROJ + (D_INNER + CONV_DIM) + h] + dt_bias[h];
    dtv[i] = (raw > 20.f) ? raw : log1pf(__expf(raw));
}

// ---------------------------------------------------------------------------
// 5) SSD chunked scan: block = (head, batch), 256 threads (8 waves).
//    Per chunk of Q=64 steps, three 64x64x64 WMMA GEMMs:
//      G = C@B^T -> M[t,s] = exp(la[t]-la[s]) * G[t,s]  (s<=t)
//      Y = M@U + (cum . C)@H0^T + D*x
//      H = exp(la[Q-1])*H + ((w . U)^T)@B ,  w[s]=exp(la[Q-1]-la[s])
// ---------------------------------------------------------------------------
__global__ void __launch_bounds__(256)
k_ssd(const float* __restrict__ xc, const float* __restrict__ dtv,
      const float* __restrict__ A_log, const float* __restrict__ Dp,
      float* __restrict__ yb) {
    const int h = blockIdx.x;
    const int b = blockIdx.y;
    const int tid  = threadIdx.x;
    const int lane = tid & 31;
    const int wave = tid >> 5;
    const int half = lane >> 4;
    const int lid  = lane & 15;
    const int kbase = half * 8;

    __shared__ _Float16 sB[QCHUNK][D_STATE];   // B_t[n]
    __shared__ _Float16 sC[QCHUNK][D_STATE];   // C_t[n]
    __shared__ _Float16 sU[QCHUNK][HEADDIM];   // dt_t * x_t[p]
    __shared__ _Float16 sM[QCHUNK][QCHUNK];    // decay-masked C B^T
    __shared__ float    sH[HEADDIM][D_STATE];  // carried state H[p][n]
    __shared__ float    sdt[QCHUNK], sLA[QCHUNK], sCUM[QCHUNK], sW[QCHUNK];

    const float eA = __expf(A_log[h]);
    const float Dh = Dp[h];

    for (int i = tid; i < HEADDIM * D_STATE; i += 256)
        sH[i >> 6][i & 63] = 0.f;
    __syncthreads();

    for (int ch = 0; ch < NCHUNK; ++ch) {
        const size_t mbase = (size_t)(b * SEQLEN + ch * QCHUNK);

        // ---- stage chunk operands ----
        for (int i = tid; i < QCHUNK * D_STATE; i += 256) {
            const int t = i >> 6, j = i & 63;
            const size_t row = (mbase + t) * CONV_DIM;
            sB[t][j] = (_Float16)xc[row + D_INNER + j];
            sC[t][j] = (_Float16)xc[row + D_INNER + D_STATE + j];
            sU[t][j] = (_Float16)(dtv[(mbase + t) * NHEADS + h] *
                                  xc[row + h * HEADDIM + j]);
        }
        if (tid < QCHUNK) sdt[tid] = dtv[(mbase + tid) * NHEADS + h];
        if (ch + 1 < NCHUNK)
            __builtin_prefetch(&xc[(mbase + QCHUNK) * CONV_DIM + D_INNER + tid], 0, 1);
        __syncthreads();

        // ---- log-decay prefix + tables ----
        if (tid == 0) {
            float run = 0.f;
            for (int t = 0; t < QCHUNK; ++t) { run -= eA * sdt[t]; sLA[t] = run; }
        }
        __syncthreads();
        if (tid < QCHUNK) {
            sCUM[tid] = __expf(sLA[tid]);
            sW[tid]   = __expf(sLA[QCHUNK - 1] - sLA[tid]);
        }
        __syncthreads();

        // ---- GEMM1: G = C @ B^T, fused into decay-masked M ----
#pragma unroll
        for (int it = 0; it < 2; ++it) {
            const int tile = wave + 8 * it;
            const int m0 = (tile >> 2) * 16, n0 = (tile & 3) * 16;
            v8f acc = {};
            acc = mm64([&](int r, int k) { return (float)sC[m0 + r][k]; },
                       [&](int k, int c) { return (float)sB[n0 + c][k]; },
                       acc, lid, kbase);
#pragma unroll
            for (int r = 0; r < 8; ++r) {
                const int t = m0 + r + half * 8;
                const int s = n0 + lid;
                const float ratio = (s <= t) ? __expf(sLA[t] - sLA[s]) : 0.f;
                sM[t][s] = (_Float16)(ratio * acc[r]);
            }
        }
        __syncthreads();

        // ---- GEMM2: Y = M@U + (cum.C)@H0^T + D*x ----
#pragma unroll
        for (int it = 0; it < 2; ++it) {
            const int tile = wave + 8 * it;
            const int m0 = (tile >> 2) * 16, n0 = (tile & 3) * 16;
            v8f acc = {};
            acc = mm64([&](int r, int k) { return (float)sM[m0 + r][k]; },
                       [&](int k, int c) { return (float)sU[k][n0 + c]; },
                       acc, lid, kbase);
            acc = mm64([&](int r, int k) { return sCUM[m0 + r] * (float)sC[m0 + r][k]; },
                       [&](int k, int c) { return sH[n0 + c][k]; },
                       acc, lid, kbase);
#pragma unroll
            for (int r = 0; r < 8; ++r) {
                const int t = m0 + r + half * 8;
                const int p = n0 + lid;
                const float xv = (float)sU[t][p] / sdt[t];   // recover x = u/dt
                yb[(mbase + t) * D_INNER + h * HEADDIM + p] = acc[r] + Dh * xv;
            }
        }
        __syncthreads();

        // ---- GEMM3: H = exp(laQ)*H + (w.U)^T @ B ----
        const float cq = sCUM[QCHUNK - 1];
#pragma unroll
        for (int it = 0; it < 2; ++it) {
            const int tile = wave + 8 * it;
            const int m0 = (tile >> 2) * 16, n0 = (tile & 3) * 16;
            v8f acc = {};
            acc = mm64([&](int r, int k) { return sW[k] * (float)sU[k][m0 + r]; },
                       [&](int k, int c) { return (float)sB[k][n0 + c]; },
                       acc, lid, kbase);
#pragma unroll
            for (int r = 0; r < 8; ++r) {
                const int p = m0 + r + half * 8;
                const int n = n0 + lid;
                sH[p][n] = cq * sH[p][n] + acc[r];
            }
        }
        __syncthreads();
    }
}

// ---------------------------------------------------------------------------
// 6) per-(b,l) RMS statistic of g = y * silu(z)
// ---------------------------------------------------------------------------
__global__ void k_rstd(const float* __restrict__ yb, const float* __restrict__ zx,
                       float* __restrict__ rstd) {
    const int m = blockIdx.x;
    __shared__ float red[256];
    float ss = 0.f;
    for (int d = threadIdx.x; d < D_INNER; d += 256) {
        const float g = yb[(size_t)m * D_INNER + d] * silu_f(zx[(size_t)m * D_IN_PROJ + d]);
        ss += g * g;
    }
    red[threadIdx.x] = ss;
    __syncthreads();
    for (int s = 128; s > 0; s >>= 1) {
        if (threadIdx.x < s) red[threadIdx.x] += red[threadIdx.x + s];
        __syncthreads();
    }
    if (threadIdx.x == 0) rstd[m] = rsqrtf(red[0] / D_INNER + EPSF);
}

// ---------------------------------------------------------------------------
// 7) gsum[b][d] = mean_l g_norm[b,l,d]   (Wout pulled past the mean)
// ---------------------------------------------------------------------------
__global__ void k_gsum(const float* __restrict__ yb, const float* __restrict__ zx,
                       const float* __restrict__ rstd, const float* __restrict__ norm_w,
                       float* __restrict__ gsum) {
    const int d = blockIdx.x * 256 + threadIdx.x;
    const int b = blockIdx.y;
    float acc = 0.f;
    for (int l = 0; l < SEQLEN; ++l) {
        const int m = b * SEQLEN + l;
        const float g = yb[(size_t)m * D_INNER + d] * silu_f(zx[(size_t)m * D_IN_PROJ + d]);
        acc += g * rstd[m];
    }
    gsum[b * D_INNER + d] = acc * norm_w[d] * (1.f / SEQLEN);
}

// ---------------------------------------------------------------------------
// 8) pooled[b][j] = gsum[b][:] @ Wout[:, j]
// ---------------------------------------------------------------------------
__global__ void k_pooled(const float* __restrict__ gsum, const float* __restrict__ Wout,
                         float* __restrict__ pooled) {
    const int j = blockIdx.x * 256 + threadIdx.x;
    const int b = blockIdx.y;
    float acc = 0.f;
    for (int d = 0; d < D_INNER; ++d)
        acc += gsum[b * D_INNER + d] * Wout[(size_t)d * D_MODEL + j];
    pooled[b * D_MODEL + j] = acc;
}

// ---------------------------------------------------------------------------
// 9) MLP head: h = pooled@W1+b1, batchnorm over batch, relu, @W2+b2
// ---------------------------------------------------------------------------
__global__ void __launch_bounds__(512)
k_head(const float* __restrict__ pooled, const float* __restrict__ W1,
       const float* __restrict__ b1, const float* __restrict__ gamma,
       const float* __restrict__ beta, const float* __restrict__ W2,
       const float* __restrict__ b2, float* __restrict__ out) {
    __shared__ float sh[BATCH][64];
    __shared__ float sv[BATCH][64];
    const int tid = threadIdx.x;
    const int b = tid >> 6, j = tid & 63;
    float acc = b1[j];
    for (int d = 0; d < D_MODEL; ++d)
        acc += pooled[b * D_MODEL + d] * W1[d * 64 + j];
    sh[b][j] = acc;
    __syncthreads();
    float mu = 0.f;
    for (int bb = 0; bb < BATCH; ++bb) mu += sh[bb][j];
    mu *= (1.f / BATCH);
    float var = 0.f;
    for (int bb = 0; bb < BATCH; ++bb) { const float d2 = sh[bb][j] - mu; var += d2 * d2; }
    var *= (1.f / BATCH);
    float hn = (acc - mu) * rsqrtf(var + EPSF) * gamma[j] + beta[j];
    hn = fmaxf(hn, 0.f);
    sv[b][j] = hn * W2[j];
    __syncthreads();
    if (tid < BATCH) {
        float s = b2[0];
        for (int jj = 0; jj < 64; ++jj) s += sv[tid][jj];
        out[tid] = s;
    }
}

// ---------------------------------------------------------------------------
extern "C" void kernel_launch(void* const* d_in, const int* in_sizes, int n_in,
                              void* d_out, int out_size, void* d_ws, size_t ws_size,
                              hipStream_t stream) {
    const float* x       = (const float*)d_in[0];
    const float* Wproj   = (const float*)d_in[1];
    const float* bproj   = (const float*)d_in[2];
    const float* Win     = (const float*)d_in[3];
    const float* conv_w  = (const float*)d_in[4];
    const float* conv_b  = (const float*)d_in[5];
    const float* dt_bias = (const float*)d_in[6];
    const float* A_log   = (const float*)d_in[7];
    const float* Dp      = (const float*)d_in[8];
    const float* norm_w  = (const float*)d_in[9];
    const float* Wout    = (const float*)d_in[10];
    const float* W1      = (const float*)d_in[11];
    const float* b1      = (const float*)d_in[12];
    const float* gamma   = (const float*)d_in[13];
    const float* beta    = (const float*)d_in[14];
    const float* W2      = (const float*)d_in[15];
    const float* b2      = (const float*)d_in[16];
    float* out = (float*)d_out;

    uint8_t* ws = (uint8_t*)d_ws;
    size_t off = 0;
    auto carve = [&](size_t bytes) -> void* {
        void* p = (void*)(ws + off);
        off = (off + bytes + 255) & ~(size_t)255;
        return p;
    };
    _Float16* Wf  = (_Float16*)carve((size_t)32 * D_IN_PROJ * sizeof(_Float16));
    float* bf     = (float*)carve((size_t)D_IN_PROJ * sizeof(float));
    float* zx     = (float*)carve((size_t)M_TOTAL * D_IN_PROJ * sizeof(float));
    float* xc     = (float*)carve((size_t)M_TOTAL * CONV_DIM * sizeof(float));
    float* dtv    = (float*)carve((size_t)M_TOTAL * NHEADS * sizeof(float));
    float* yb     = (float*)carve((size_t)M_TOTAL * D_INNER * sizeof(float));
    float* rstd   = (float*)carve((size_t)M_TOTAL * sizeof(float));
    float* gsum   = (float*)carve((size_t)BATCH * D_INNER * sizeof(float));
    float* pooled = (float*)carve((size_t)BATCH * D_MODEL * sizeof(float));

    k_fuse_weights<<<dim3(D_IN_PROJ), dim3(32), 0, stream>>>(Wproj, bproj, Win, Wf, bf);
    k_gemm_in<<<dim3(D_IN_PROJ / 16, M_TOTAL / (16 * 8)), dim3(256), 0, stream>>>(x, Wf, bf, zx);
    k_conv_silu<<<dim3((CONV_DIM + 255) / 256, M_TOTAL), dim3(256), 0, stream>>>(zx, conv_w, conv_b, xc);
    k_dt<<<dim3((M_TOTAL * NHEADS + 255) / 256), dim3(256), 0, stream>>>(zx, dt_bias, dtv);
    k_ssd<<<dim3(NHEADS, BATCH), dim3(256), 0, stream>>>(xc, dtv, A_log, Dp, yb);
    k_rstd<<<dim3(M_TOTAL), dim3(256), 0, stream>>>(yb, zx, rstd);
    k_gsum<<<dim3(D_INNER / 256, BATCH), dim3(256), 0, stream>>>(yb, zx, rstd, norm_w, gsum);
    k_pooled<<<dim3(D_MODEL / 256, BATCH), dim3(256), 0, stream>>>(gsum, Wout, pooled);
    k_head<<<dim3(1), dim3(512), 0, stream>>>(pooled, W1, b1, gamma, beta, W2, b2, out);
}
